// TFRNN_65893388255688
// MI455X (gfx1250) — matrix operand order repaired and checked by reading, not compiled, not run on hardware
//
#include <hip/hip_runtime.h>
#include <cmath>

// ---------------------------------------------------------------------------
// Persistent-kernel RNN for MI455X (gfx1250, wave32, WMMA).
//   h_t = tanh(h_{t-1} @ W_hh^T + [x_t @ W_ih^T + b_ih + b_hh])
//   out = relu(h @ W1^T + b1) @ W2^T + b2
// The input projection has no recurrent dependency: it is precomputed for all
// 512 steps by a parallel WMMA kernel (xprep) into an f16 buffer, so the
// serial persistent loop is a pure 64x1024x1024 f16 WMMA GEMM + tanh per step.
// 16 workgroups (one per WGP), each owns 64 hidden columns; W_hh lives in LDS
// as f16; h ping-pongs in global f16 buffers; one grid barrier per step.
// ---------------------------------------------------------------------------

typedef _Float16 half4  __attribute__((ext_vector_type(4)));
typedef _Float16 half8  __attribute__((ext_vector_type(8)));
typedef _Float16 half16 __attribute__((ext_vector_type(16)));
typedef float    float8 __attribute__((ext_vector_type(8)));

#define HID   1024
#define NIN   128
#define BATCH 64
#define SEQ   512
#define KTOT  1152          // HID + NIN
#define LDSW  1160          // padded row stride (f16) -> 580 dwords, 580%64=4
#define NBLK  16            // persistent workgroups

#if __has_builtin(__builtin_amdgcn_tanhf)
#define FAST_TANH(v) __builtin_amdgcn_tanhf(v)
#else
#define FAST_TANH(v) tanhf(v)
#endif

static __device__ __forceinline__ half16 cat88(half8 lo, half8 hi) {
  return __builtin_shufflevector(lo, hi, 0,1,2,3,4,5,6,7,8,9,10,11,12,13,14,15);
}

static __device__ __forceinline__ float8 wmma16(half16 a, half16 b, float8 c) {
  // D = A(16x32 f16) x B(32x16 f16) + C(16x16 f32)
  return __builtin_amdgcn_wmma_f32_16x16x32_f16(false, a, false, b,
                                                (short)0, c, false, false);
}

// B fragment (contiguous k order, same as the A fragment): lane holds weight row = output column,
// K = ks*32 + (lane>=16 ? 16 : 0) + 0..15, contiguous in the LDS row.
static __device__ __forceinline__ half16 load_bfrag(const _Float16* wrow,
                                                    int ks, int kb2) {
  const _Float16* p = wrow + ks * 32 + kb2;
  return cat88(*(const half8*)p, *(const half8*)(p + 8));
}

// A fragment from f16 h state: lane holds batch row;
// K = ks*32 + (lane>=16 ? 16 : 0) + 0..15.
static __device__ __forceinline__ half16 load_hfrag(const _Float16* hrow,
                                                    int ks, int kb) {
  const _Float16* hr = hrow + ks * 32 + kb;
  return cat88(*(const half8*)hr, *(const half8*)(hr + 8));
}

// A fragment from fp32 x_t, converted in-register (fallback path, xs = 0..3).
static __device__ __forceinline__ half16 load_xfrag(const float* xrow,
                                                    int xs, int kb) {
  const float* xr = xrow + xs * 32 + kb;
  half16 a;
#pragma unroll
  for (int i = 0; i < 8; ++i) {
    a[i]     = (_Float16)xr[i];
    a[i + 8] = (_Float16)xr[8 + i];
  }
  return a;
}

// Grid-wide release/acquire barrier on a monotonic counter (zeroed per launch).
static __device__ __forceinline__ void grid_barrier(unsigned* ctr, unsigned target) {
  __threadfence();
  __syncthreads();
  if (threadIdx.x == 0) {
    __hip_atomic_fetch_add(ctr, 1u, __ATOMIC_RELEASE, __HIP_MEMORY_SCOPE_AGENT);
    while (__hip_atomic_load(ctr, __ATOMIC_ACQUIRE, __HIP_MEMORY_SCOPE_AGENT) < target) {
      __builtin_amdgcn_s_sleep(2);
    }
  }
  __syncthreads();
}

// Zero barrier counter + convert h0 (fp32) -> f16 ping buffer.
__global__ void init_kernel(const float* __restrict__ h0,
                            _Float16* __restrict__ hb0,
                            unsigned* __restrict__ ctr) {
  size_t idx = (size_t)blockIdx.x * blockDim.x + threadIdx.x;
  if (idx == 0) *ctr = 0u;
  size_t stride = (size_t)gridDim.x * blockDim.x;
  for (size_t i = idx; i < (size_t)BATCH * HID; i += stride)
    hb0[i] = (_Float16)h0[i];
}

// Parallel input projection: pre[t][b][j] = x[b][t] . W_ih[j] + b_ih[j] + b_hh[j].
// One block per time step (512 blocks, fully parallel, off the serial chain).
__global__ void __launch_bounds__(128, 1)
xprep_kernel(const float* __restrict__ x, const float* __restrict__ W_ih,
             const float* __restrict__ b_ih, const float* __restrict__ b_hh,
             _Float16* __restrict__ pre) {
  const int t    = blockIdx.x;
  const int lane = threadIdx.x & 31;
  const int wv   = threadIdx.x >> 5;       // wave 0..3
  const int arow = lane & 15;
  const int kb   = (lane >> 4) * 16;
  const int kb2  = (lane >> 4) * 16;
  const int rowadd = (lane >> 4) * 8;

  for (int q = 0; q < 16; ++q) {
    const int nt = wv * 16 + q;            // n-tile 0..63
    const int n  = nt * 16 + arow;
    const float bias = b_ih[n] + b_hh[n];
    half16 bf[4];
#pragma unroll
    for (int ks = 0; ks < 4; ++ks) {
      const float* wr = W_ih + (size_t)n * NIN + ks * 32 + kb2;
#pragma unroll
      for (int i = 0; i < 16; ++i) bf[ks][i] = (_Float16)wr[i];
    }
#pragma unroll
    for (int m = 0; m < 4; ++m) {
      float8 acc = (float8){bias, bias, bias, bias, bias, bias, bias, bias};
      const float* xrow = x + ((size_t)(m * 16 + arow) * SEQ + t) * NIN;
#pragma unroll
      for (int ks = 0; ks < 4; ++ks)
        acc = wmma16(load_xfrag(xrow, ks, kb), bf[ks], acc);
#pragma unroll
      for (int r = 0; r < 8; ++r) {
        const int b = m * 16 + rowadd + r;
        pre[((size_t)t * BATCH + b) * HID + n] = (_Float16)acc[r];
      }
    }
  }
}

// Persistent recurrent kernel: 16 blocks x 256 threads (8 waves).
// Wave wv: m-tile = wv>>1, n-tiles = {(wv&1)*32, (wv&1)*32+16} local columns.
// mode==1: input projection precomputed in `pre`; mode==0: fp32 x in-loop.
__global__ void __launch_bounds__(256, 1)
rnn_kernel(const float* __restrict__ x,
           const float* __restrict__ W_ih, const float* __restrict__ b_ih,
           const float* __restrict__ W_hh, const float* __restrict__ b_hh,
           _Float16* __restrict__ hb0, _Float16* __restrict__ hb1,
           const _Float16* __restrict__ pre, int mode,
           unsigned* __restrict__ ctr, float* __restrict__ hlast_out) {
  extern __shared__ char smem[];
  _Float16* wlds = (_Float16*)smem;                               // [64][LDSW]
  float*    blds = (float*)(smem + 64 * LDSW * sizeof(_Float16)); // [64]

  const int tid = threadIdx.x;
  const int g   = blockIdx.x;
  const int C0  = g * 64;              // first global hidden column of this WG

  // --- one-time: stage this WG's weight slice into LDS as f16 ---------------
  for (int idx = tid; idx < 64 * (KTOT / 4); idx += 256) {
    const int r  = idx / (KTOT / 4);
    const int k4 = (idx - r * (KTOT / 4)) * 4;
    float4 v = (k4 < HID) ? *(const float4*)(W_hh + (size_t)(C0 + r) * HID + k4)
                          : *(const float4*)(W_ih + (size_t)(C0 + r) * NIN + (k4 - HID));
    *(half4*)(wlds + (size_t)r * LDSW + k4) =
        (half4){(_Float16)v.x, (_Float16)v.y, (_Float16)v.z, (_Float16)v.w};
  }
  if (tid < 64) blds[tid] = b_ih[C0 + tid] + b_hh[C0 + tid];
  __syncthreads();

  const int lane   = tid & 31;
  const int wv     = tid >> 5;            // wave 0..7
  const int arow   = lane & 15;
  const int kb     = (lane >> 4) * 16;    // A fragment K base
  const int kb2    = (lane >> 4) * 16;    // B fragment K base
  const int rowadd = (lane >> 4) * 8;     // C/D: lanes 16-31 hold M+8
  const int m      = wv >> 1;             // batch m-tile 0..3
  const int nloc0  = (wv & 1) * 32 + arow;
  const int nloc1  = nloc0 + 16;
  const int jcol0  = C0 + nloc0;
  const int jcol1  = C0 + nloc1;
  const float bias0 = blds[nloc0];
  const float bias1 = blds[nloc1];
  const _Float16* wrow0 = wlds + (size_t)nloc0 * LDSW;
  const _Float16* wrow1 = wlds + (size_t)nloc1 * LDSW;
  const int brow = m * 16 + arow;         // batch row this lane supplies to A

  for (int t = 0; t < SEQ; ++t) {
    const _Float16* hprev = (t & 1) ? hb1 : hb0;
    _Float16*       hnext = (t & 1) ? hb0 : hb1;
    const _Float16* hrow = hprev + (size_t)brow * HID;

    // Pre-activation tile: issue loads up front so they overlap the K loop.
    float pr0[8], pr1[8];
    if (mode) {
#pragma unroll
      for (int r = 0; r < 8; ++r) {
        const size_t b = (size_t)(m * 16 + rowadd + r);
        pr0[r] = (float)pre[((size_t)t * BATCH + b) * HID + jcol0];
        pr1[r] = (float)pre[((size_t)t * BATCH + b) * HID + jcol1];
      }
    } else {
#pragma unroll
      for (int r = 0; r < 8; ++r) { pr0[r] = 0.0f; pr1[r] = 0.0f; }
    }

    const float i0 = mode ? 0.0f : bias0;
    const float i1 = mode ? 0.0f : bias1;
    float8 acc0 = (float8){i0, i0, i0, i0, i0, i0, i0, i0};
    float8 acc1 = (float8){i1, i1, i1, i1, i1, i1, i1, i1};

    if (!mode) {   // fallback: x_t contribution in-loop (weight rows k>=1024)
      const float* xrow = x + ((size_t)brow * SEQ + t) * NIN;
#pragma unroll
      for (int xs = 0; xs < 4; ++xs) {
        half16 ax = load_xfrag(xrow, xs, kb);
        acc0 = wmma16(ax, load_bfrag(wrow0, 32 + xs, kb2), acc0);
        acc1 = wmma16(ax, load_bfrag(wrow1, 32 + xs, kb2), acc1);
      }
    }

    // --- h contribution: 32 branch-free k-steps, two fixed fragment sets ---
    half16 a0  = load_hfrag(hrow, 0, kb);
    half16 f00 = load_bfrag(wrow0, 0, kb2);
    half16 f10 = load_bfrag(wrow1, 0, kb2);
#pragma unroll 2
    for (int ks = 0; ks < 32; ks += 2) {
      half16 a1  = load_hfrag(hrow, ks + 1, kb);
      half16 f01 = load_bfrag(wrow0, ks + 1, kb2);
      half16 f11 = load_bfrag(wrow1, ks + 1, kb2);
      acc0 = wmma16(a0, f00, acc0);
      acc1 = wmma16(a0, f10, acc1);
      if (ks < 30) {
        a0  = load_hfrag(hrow, ks + 2, kb);
        f00 = load_bfrag(wrow0, ks + 2, kb2);
        f10 = load_bfrag(wrow1, ks + 2, kb2);
      }
      acc0 = wmma16(a1, f01, acc0);
      acc1 = wmma16(a1, f11, acc1);
    }

    const bool last = (t == SEQ - 1);
#pragma unroll
    for (int r = 0; r < 8; ++r) {
      const int b = m * 16 + rowadd + r;
      float v0 = FAST_TANH(acc0[r] + pr0[r]);
      float v1 = FAST_TANH(acc1[r] + pr1[r]);
      hnext[(size_t)b * HID + jcol0] = (_Float16)v0;
      hnext[(size_t)b * HID + jcol1] = (_Float16)v1;
      if (last) {
        hlast_out[(size_t)b * HID + jcol0] = v0;   // fp32 h_last output
        hlast_out[(size_t)b * HID + jcol1] = v1;
      }
    }
    grid_barrier(ctr, (unsigned)(NBLK * (t + 1)));
  }
}

// Head GEMM: out = act(A[64xK]_f16 @ W[NxK]_f32^T + bias).
__global__ void __launch_bounds__(128, 1)
head_gemm(const _Float16* __restrict__ A, const float* __restrict__ W,
          const float* __restrict__ bias, int K, int ld, int relu,
          _Float16* __restrict__ out16, float* __restrict__ out32) {
  const int tid  = threadIdx.x;
  const int lane = tid & 31;
  const int wv   = tid >> 5;
  const int arow = lane & 15;
  const int kb   = (lane >> 4) * 16;
  const int kb2  = (lane >> 4) * 16;
  const int n    = (blockIdx.x * 4 + wv) * 16 + arow;
  const float bv = bias[n];

  float8 acc[4];
#pragma unroll
  for (int mm = 0; mm < 4; ++mm)
    acc[mm] = (float8){bv, bv, bv, bv, bv, bv, bv, bv};

  const float* wr = W + (size_t)n * K;
  const int nks = K / 32;
  for (int ks = 0; ks < nks; ++ks) {
    const int k0 = ks * 32;
    half16 bf;
#pragma unroll
    for (int i = 0; i < 16; ++i) bf[i] = (_Float16)wr[k0 + kb2 + i];
#pragma unroll
    for (int mm = 0; mm < 4; ++mm) {
      const _Float16* ar = A + (size_t)(mm * 16 + arow) * K + k0 + kb;
      half16 af = cat88(*(const half8*)ar, *(const half8*)(ar + 8));
      acc[mm] = wmma16(af, bf, acc[mm]);
    }
  }

  const int rowadd = (lane >> 4) * 8;
#pragma unroll
  for (int mm = 0; mm < 4; ++mm) {
#pragma unroll
    for (int r = 0; r < 8; ++r) {
      float v = acc[mm][r];
      if (relu) v = fmaxf(v, 0.0f);
      int b = mm * 16 + rowadd + r;
      if (out16) out16[(size_t)b * ld + n] = (_Float16)v;
      if (out32) out32[(size_t)b * ld + n] = v;
    }
  }
}

extern "C" void kernel_launch(void* const* d_in, const int* in_sizes, int n_in,
                              void* d_out, int out_size, void* d_ws, size_t ws_size,
                              hipStream_t stream) {
  const float* x    = (const float*)d_in[0];
  const float* h0   = (const float*)d_in[1];
  const float* W_ih = (const float*)d_in[2];
  const float* b_ih = (const float*)d_in[3];
  const float* W_hh = (const float*)d_in[4];
  const float* b_hh = (const float*)d_in[5];
  const float* W1   = (const float*)d_in[6];
  const float* b1   = (const float*)d_in[7];
  const float* W2   = (const float*)d_in[8];
  const float* b2   = (const float*)d_in[9];
  float* out = (float*)d_out;   // [64*128 outputs_o | 64*1024 h_last]

  // Workspace: counter | h ping | h pong | mlp | pre (f16 buffers).
  char* ws = (char*)d_ws;
  unsigned* ctr  = (unsigned*)ws;
  _Float16* hb0  = (_Float16*)(ws + 256);
  _Float16* hb1  = hb0 + (size_t)BATCH * HID;
  _Float16* mlp  = hb1 + (size_t)BATCH * HID;
  _Float16* pre  = mlp + (size_t)BATCH * HID;

  const size_t preNeed = 256 + (size_t)3 * BATCH * HID * 2
                       + (size_t)SEQ * BATCH * HID * 2;   // ~67.5 MB
  const int mode = (ws_size >= preNeed) ? 1 : 0;          // fixed per harness

  const size_t smem = (size_t)64 * LDSW * sizeof(_Float16) + 64 * sizeof(float); // ~149 KB
  (void)hipFuncSetAttribute((const void*)rnn_kernel,
                            hipFuncAttributeMaxDynamicSharedMemorySize, (int)smem);

  init_kernel<<<64, 256, 0, stream>>>(h0, hb0, ctr);
  if (mode)
    xprep_kernel<<<SEQ, 128, 0, stream>>>(x, W_ih, b_ih, b_hh, pre);
  rnn_kernel<<<NBLK, 256, smem, stream>>>(x, W_ih, b_ih, W_hh, b_hh,
                                          hb0, hb1, pre, mode, ctr,
                                          out + BATCH * NIN);
  // 512 steps end writing hb0 (511 is odd -> hnext = hb0).
  head_gemm<<<16, 128, 0, stream>>>(hb0, W1, b1, HID, HID, 1, mlp, nullptr);
  head_gemm<<<2, 128, 0, stream>>>(mlp, W2, b2, HID, NIN, 0, nullptr, out);
}
